// WedgeClassifier0_22411139350995
// MI455X (gfx1250) — matrix operand order repaired and checked by reading, not compiled, not run on hardware
//
#include <hip/hip_runtime.h>

// ---------------------------------------------------------------------------
// WedgeClassifier on MI455X (gfx1250, wave32, WMMA).
//
// out = (segment_sum(x, seg) @ W2) + B2   where W2/B2 fold FGL+FC weights.
// Stage 1 is a one-hot GEMM: sums = X(1024x65536) @ OneHot(seg)(65536x16pad).
// Memory-bound: 256 MB of x @ 23.3 TB/s ~= 11 us floor; WMMA keeps compute
// ~0.1 instr/element so HBM stays the binding constraint (scalar select-add
// would be VALU-bound at ~2x the memory time).
// ---------------------------------------------------------------------------

#define PPIX    65536            // pixels per image (256*256)
#define NBATCH  1024
#define CHUNK   1024             // pixels staged in LDS per iteration
#define SH      (CHUNK + 8)      // LDS row stride in halves (pad: bank spread)
#define KSPLIT  16               // K-dimension split across blocks
#define KSLICE  (PPIX / KSPLIT)  // 4096 pixels per block
#define NCHUNK  (KSLICE / CHUNK) // 4
#define TILESM  (NBATCH / 16)    // 64 row tiles

typedef _Float16 v16h __attribute__((ext_vector_type(16)));
typedef _Float16 v8h  __attribute__((ext_vector_type(8)));
typedef _Float16 v4h  __attribute__((ext_vector_type(4)));
typedef float    v8f  __attribute__((ext_vector_type(8)));

// ---------------------------------------------------------------------------
// Kernel 0: fold fgl_w/fgl_b/fc_w/fc_b into W2 (16x16, zero padded) and B2(16)
//   W2[g,k] = sum_c fgl_w[g,0,c] * fc_w[k, c*9+g]          (g<9, k<8)
//   B2[k]   = fc_b[k] + sum_{g,c} fgl_b[g,c] * fc_w[k, c*9+g]
// ---------------------------------------------------------------------------
__global__ __launch_bounds__(256) void prep_kernel(
    const float* __restrict__ fgl_w, const float* __restrict__ fgl_b,
    const float* __restrict__ fc_w,  const float* __restrict__ fc_b,
    float* __restrict__ W2, float* __restrict__ B2) {
  const int t = threadIdx.x;
  {
    const int g = t >> 4, k = t & 15;
    float v = 0.0f;
    if (g < 9 && k < 8) {
      #pragma unroll
      for (int c = 0; c < 4; ++c)
        v += fgl_w[g * 4 + c] * fc_w[k * 36 + c * 9 + g];
    }
    W2[t] = v;
  }
  if (t < 16) {
    float v = 0.0f;
    if (t < 8) {
      v = fc_b[t];
      for (int g = 0; g < 9; ++g)
        #pragma unroll
        for (int c = 0; c < 4; ++c)
          v += fgl_b[g * 4 + c] * fc_w[t * 36 + c * 9 + g];
    }
    B2[t] = v;
  }
}

// ---------------------------------------------------------------------------
// Kernel 1: one-hot segment-sum via v_wmma_f32_16x16x32_f16.
// Grid (64, 16): blockIdx.x = 16-row tile, blockIdx.y = K split.
// Each block streams 16 rows x 4096 pixels (256 KB) through LDS (f16),
// 8 waves each accumulate a 16x16 f32 C over their K share, then a
// deterministic LDS tree-reduction writes per-(tile,ksplit) partials.
// ---------------------------------------------------------------------------
__global__ __launch_bounds__(256) void segsum_kernel(
    const float* __restrict__ x, const int* __restrict__ seg,
    float* __restrict__ part) {
  __shared__ alignas(16) unsigned char smem[16 * SH * 2 + CHUNK * 4];
  _Float16* xt  = (_Float16*)smem;                  // 16 x SH halves
  int*      sg  = (int*)(smem + 16 * SH * 2);       // CHUNK seg ids
  float*    red = (float*)smem;                     // aliased: 8 x 256 floats

  const int t    = threadIdx.x;
  const int lane = t & 31;
  const int wave = t >> 5;
  const int r0   = blockIdx.x * 16;
  const int ks   = blockIdx.y;

  const int mn    = lane & 15;        // A row (M) == B column (N) for this lane
  const int hsel  = lane >> 4;        // lane half
  const int koffA = 8  * hsel;        // ISA 16-bit A layout: K runs koffA..+7, 16+koffA..+7
  const int kbasB = 8  * hsel;        // B layout matches A: K runs kbasB..+7, 16+kbasB..+7
  const _Float16 one  = (_Float16)1.0f;
  const _Float16 zero = (_Float16)0.0f;

  v8f acc = {};

  for (int ch = 0; ch < NCHUNK; ++ch) {
    const int kb = ks * KSLICE + ch * CHUNK;
    __syncthreads();  // previous chunk's readers done before restaging

    // Stage x chunk: row i, 256 threads x 1 float4 -> coalesced 512B/wave.
    #pragma unroll 4
    for (int i = 0; i < 16; ++i) {
      const float* xp = x + (size_t)(r0 + i) * PPIX + kb + t * 4;
      const float4 v = *(const float4*)xp;
      if (ch + 1 < NCHUNK) __builtin_prefetch(xp + CHUNK, 0, 0);  // global_prefetch_b8
      v4h h;
      h[0] = (_Float16)v.x; h[1] = (_Float16)v.y;
      h[2] = (_Float16)v.z; h[3] = (_Float16)v.w;
      *(v4h*)(xt + i * SH + t * 4) = h;
    }
    // Stage seg chunk (wave-shared, reused by all 64 row tiles via L2 anyway).
    *(int4*)(sg + t * 4) = *(const int4*)(seg + kb + t * 4);
    __syncthreads();

    // 32 K-steps of 32 pixels per chunk; wave w owns steps [4w, 4w+4).
    #pragma unroll
    for (int s = 0; s < 4; ++s) {
      const int k0 = wave * 128 + s * 32;

      // A tile (16x32 f16): lane = row mn; two 16B runs per ISA layout.
      const _Float16* xr = xt + mn * SH + k0 + koffA;
      const v8h lo = *(const v8h*)(xr);
      const v8h hi = *(const v8h*)(xr + 16);
      const v16h a = __builtin_shufflevector(
          lo, hi, 0, 1, 2, 3, 4, 5, 6, 7, 8, 9, 10, 11, 12, 13, 14, 15);

      // B tile (32x16 f16) = one-hot(seg): lane = column mn, K runs kbasB..+7, 16+kbasB..+7.
      const int* sp = sg + k0 + kbasB;
      const int4 s0 = *(const int4*)(sp);
      const int4 s1 = *(const int4*)(sp + 4);
      const int4 s2 = *(const int4*)(sp + 16);
      const int4 s3 = *(const int4*)(sp + 20);
      const int sv[16] = {s0.x, s0.y, s0.z, s0.w, s1.x, s1.y, s1.z, s1.w,
                          s2.x, s2.y, s2.z, s2.w, s3.x, s3.y, s3.z, s3.w};
      v16h b;
      #pragma unroll
      for (int h = 0; h < 16; ++h) b[h] = (sv[h] == mn) ? one : zero;

      acc = __builtin_amdgcn_wmma_f32_16x16x32_f16(
          false, a, false, b, (short)0, acc, false, false);
    }
  }

  // Deterministic cross-wave reduction of the 8 partial C tiles.
  __syncthreads();
  #pragma unroll
  for (int r = 0; r < 8; ++r) {
    const int M = r + 8 * hsel;                    // ISA f32 C/D layout
    red[wave * 256 + M * 16 + mn] = acc[r];
  }
  __syncthreads();
  float ssum = 0.0f;
  #pragma unroll
  for (int w = 0; w < 8; ++w) ssum += red[w * 256 + t];
  part[((size_t)blockIdx.x * KSPLIT + ks) * 256 + t] = ssum;
}

// ---------------------------------------------------------------------------
// Kernel 2: reduce K-split partials, then one WMMA per tile: out = sums@W2+B2.
// K = 16 groups, zero-padded to 32 in the f16 operands.
// ---------------------------------------------------------------------------
__global__ __launch_bounds__(256) void final_kernel(
    const float* __restrict__ part, const float* __restrict__ W2,
    const float* __restrict__ B2, float* __restrict__ out) {
  __shared__ float ssum[256];
  __shared__ float sb2[16];
  const int t    = threadIdx.x;
  const int tile = blockIdx.x;

  float a0 = 0.0f;
  #pragma unroll
  for (int j = 0; j < KSPLIT; ++j)
    a0 += part[((size_t)tile * KSPLIT + j) * 256 + t];
  ssum[t] = a0;
  if (t < 16) sb2[t] = B2[t];
  __syncthreads();

  if (t < 32) {  // wave 0, full EXEC within the wave
    const int lane  = t;
    const int mn    = lane & 15;
    const int hsel  = lane >> 4;
    const int koffA = 8 * hsel;
    const int kbasB = 8 * hsel;
    v16h a, b;
    #pragma unroll
    for (int h = 0; h < 16; ++h) {
      const int Ka = (h < 8) ? (koffA + h) : (16 + koffA + (h - 8));
      a[h] = (Ka < 16) ? (_Float16)ssum[mn * 16 + Ka] : (_Float16)0.0f;
      const int Kb = (h < 8) ? (kbasB + h) : (16 + kbasB + (h - 8));
      b[h] = (Kb < 16) ? (_Float16)W2[Kb * 16 + mn] : (_Float16)0.0f;
    }
    v8f c = {};
    c = __builtin_amdgcn_wmma_f32_16x16x32_f16(
        false, a, false, b, (short)0, c, false, false);
    #pragma unroll
    for (int r = 0; r < 8; ++r) {
      const int M = r + 8 * hsel;
      if (mn < 8) out[(size_t)(tile * 16 + M) * 8 + mn] = c[r] + sb2[mn];
    }
  }
}

// ---------------------------------------------------------------------------
extern "C" void kernel_launch(void* const* d_in, const int* in_sizes, int n_in,
                              void* d_out, int out_size, void* d_ws,
                              size_t ws_size, hipStream_t stream) {
  (void)in_sizes; (void)n_in; (void)out_size; (void)ws_size;
  const float* x     = (const float*)d_in[0];
  const int*   seg   = (const int*)d_in[1];
  const float* fgl_w = (const float*)d_in[2];
  const float* fgl_b = (const float*)d_in[3];
  const float* fc_w  = (const float*)d_in[4];
  const float* fc_b  = (const float*)d_in[5];
  float* out = (float*)d_out;

  float* W2   = (float*)d_ws;        // 256 floats
  float* B2   = W2 + 256;            // 16 floats
  float* part = W2 + 512;            // 64*16*256 floats = 1 MB

  prep_kernel<<<1, 256, 0, stream>>>(fgl_w, fgl_b, fc_w, fc_b, W2, B2);
  segsum_kernel<<<dim3(TILESM, KSPLIT), 256, 0, stream>>>(x, seg, part);
  final_kernel<<<TILESM, 256, 0, stream>>>(part, W2, B2, out);
}
